// MiaoBlock_20847771255434
// MI455X (gfx1250) — matrix operand order repaired and checked by reading, not compiled, not run on hardware
//
#include <hip/hip_runtime.h>
#include <math.h>

// ---------------------------------------------------------------------------
// MI455X (gfx1250) implementation of the MACE-style equivariant block.
// All (E*3^o,64)x(64,64) channel-mix GEMMs and the gate matmuls run on
// v_wmma_f32_16x16x32_bf16 (bf16 in, f32 accumulate). Global data stays f32.
// This revision caps kernels at 256 VGPRs to kill s_set_vgpr_msb overhead and
// keep multi-wave occupancy per SIMD32.
// ---------------------------------------------------------------------------

typedef __bf16 bf16_t;
typedef __attribute__((ext_vector_type(16))) __bf16 v16bf;
typedef __attribute__((ext_vector_type(8)))  __bf16 v8bf;
typedef __attribute__((ext_vector_type(8)))  float  v8f;

#define WAVES 3
#define BLK   (WAVES * 32)
#define VGPR_CAP __attribute__((amdgpu_num_vgpr(256)))

__device__ __forceinline__ float siluf(float x) { return x * (1.0f / (1.0f + __expf(-x))); }

// branchless 3-way select (v_cndmask, avoids scratch from dynamic arrays)
__device__ __forceinline__ float sel3(int a, float x0, float x1, float x2) {
  float r = x0;
  r = (a == 1) ? x1 : r;
  r = (a == 2) ? x2 : r;
  return r;
}

__device__ __forceinline__ v16bf join16(v8bf a, v8bf b) {
  return __builtin_shufflevector(a, b, 0,1,2,3,4,5,6,7,8,9,10,11,12,13,14,15);
}
// A-fragment: 8 bf16 at p, 8 bf16 at p+16 (16-bit A 16x32 lane layout)
__device__ __forceinline__ v16bf loadApair(const bf16_t* p) {
  v8bf lo = *(const v8bf*)p;
  v8bf hi = *(const v8bf*)(p + 16);
  return join16(lo, hi);
}
// 16 contiguous bf16 (B-fragment channel run: K = half*16 + q)
__device__ __forceinline__ v16bf loadRun16(const bf16_t* p) {
  v8bf lo = *(const v8bf*)p;
  v8bf hi = *(const v8bf*)(p + 8);
  return join16(lo, hi);
}
// B = (f .* z) * s  elementwise over a 16-channel run, rounded to bf16
__device__ __forceinline__ v16bf mulFrag(const bf16_t* z, const bf16_t* f, float s) {
  v16bf Z = loadRun16(z), F = loadRun16(f), out;
#pragma unroll
  for (int q = 0; q < 16; ++q) out[q] = (bf16_t)((float)Z[q] * (float)F[q] * s);
  return out;
}
__device__ __forceinline__ void zero4(v8f* a) {
#pragma unroll
  for (int t = 0; t < 4; ++t) {
#pragma unroll
    for (int r = 0; r < 8; ++r) a[t][r] = 0.0f;
  }
}
// D(64d x 16e) += Wt-tile(64d x 64c) x B(64c x 16e) using 8x wmma bf16
__device__ __forceinline__ void gemmAcc(v8f* acc, const bf16_t* W, v16bf B0, v16bf B1, int lane) {
  const int m = lane & 15, h = lane >> 4;
#pragma unroll
  for (int t = 0; t < 4; ++t) {
    const bf16_t* row = W + (size_t)(t * 16 + m) * 64 + h * 16;
    v16bf A0 = loadRun16(row);
    v16bf A1 = loadRun16(row + 32);
    acc[t] = __builtin_amdgcn_wmma_f32_16x16x32_bf16(false, A0, false, B0, (short)0, acc[t], false, false);
    acc[t] = __builtin_amdgcn_wmma_f32_16x16x32_bf16(false, A1, false, B1, (short)0, acc[t], false, false);
  }
}

// Per-wave LDS tile: z-arrays channel-contiguous (bf16) + rbf + u + idx_i
struct __align__(16) WaveTile {
  bf16_t sb0[16][64];      // base0
  bf16_t sb1[16][3][64];   // base1[a]
  bf16_t sb2[16][9][64];   // base2[a*3+b]
  bf16_t sv2[16][3][64];   // v2[a] = sum_b base2[a][b]*u[b]
  bf16_t ss1[16][64];      // base1 . u
  bf16_t ss2[16][64];      // u . base2 . u
  bf16_t fbuf[4][16][64];  // f_k for current o-group (<=4 combos)
  float  srbf[16][12];
  float  uvec[16][3];
  int    eidx[16];
};

// ---------------------------------------------------------------------------
// small prep kernels
// ---------------------------------------------------------------------------
__global__ void kZero(float* p, long n) {
  long i = (long)blockIdx.x * blockDim.x + threadIdx.x;
  if (i < n) p[i] = 0.0f;
}

__global__ void kPrepEdge(const float* __restrict__ rij, float* __restrict__ rbf,
                          float* __restrict__ rhat, int E) {
  int e = blockIdx.x * blockDim.x + threadIdx.x;
  if (e >= E) return;
  float x = rij[e * 3 + 0], y = rij[e * 3 + 1], z = rij[e * 3 + 2];
  float d = sqrtf(x * x + y * y + z * z);
  float inv = 1.0f / (d + 1e-9f);
  rhat[e * 3 + 0] = x * inv; rhat[e * 3 + 1] = y * inv; rhat[e * 3 + 2] = z * inv;
  float dc = fminf(fmaxf(d, 0.0f), 5.0f);
  float fc = 0.5f * (__cosf(3.14159265358979f * dc / 5.0f) + 1.0f);
  const float width = 5.0f / 12.0f;
#pragma unroll
  for (int b = 0; b < 12; ++b) {
    float cen = 5.0f * (float)b / 11.0f;
    float t = (d - cen) / width;
    rbf[(size_t)e * 12 + b] = __expf(-t * t) * fc;
  }
}

// transpose [k][c][d] f32 -> [k][d][c] bf16 (Wmix / Wg)
__global__ void kCvtT(const float* __restrict__ src, bf16_t* __restrict__ dst, int nmat) {
  int i = blockIdx.x * blockDim.x + threadIdx.x;
  if (i >= nmat * 4096) return;
  int k = i >> 12, rem = i & 4095, d = rem >> 6, c = rem & 63;
  dst[i] = (bf16_t)src[(k << 12) + (c << 6) + d];
}
// straight f32 -> bf16 (Wrbf, layout kept [k][b][c])
__global__ void kCvt(const float* __restrict__ src, bf16_t* __restrict__ dst, int n) {
  int i = blockIdx.x * blockDim.x + threadIdx.x;
  if (i < n) dst[i] = (bf16_t)src[i];
}

// node gate-input norms: nnorm[n][0:64] = sum_a agg1^2, nnorm[n][64:128] = sum_ab agg2^2
__global__ void kNodeNorm(const float* __restrict__ agg1, const float* __restrict__ agg2,
                          float* __restrict__ nnorm, int N) {
  int i = blockIdx.x * blockDim.x + threadIdx.x;
  if (i >= N * 64) return;
  int n = i >> 6, c = i & 63;
  float s1 = 0.0f, s2 = 0.0f;
#pragma unroll
  for (int s = 0; s < 3; ++s) { float v = agg1[(size_t)i * 3 + s]; s1 += v * v; }
#pragma unroll
  for (int s = 0; s < 9; ++s) { float v = agg2[(size_t)i * 9 + s]; s2 += v * v; }
  nnorm[(size_t)n * 128 + c]      = s1;
  nnorm[(size_t)n * 128 + 64 + c] = s2;
}

// ---------------------------------------------------------------------------
// main message kernel. MODE 0: node conv (atomic segment-sum into agg, /16).
//                     MODE 1: edge conv (raw msgs + norms, o=0 finalized).
// ---------------------------------------------------------------------------
template <int MODE>
__global__ VGPR_CAP __launch_bounds__(BLK)
void kMsg(const float* __restrict__ node0, const float* __restrict__ node1,
          const float* __restrict__ node2, const float* __restrict__ edge0,
          const float* __restrict__ edge1, const float* __restrict__ edge2,
          const int* __restrict__ idx_i, const int* __restrict__ idx_j,
          const float* __restrict__ rbf, const float* __restrict__ rhat,
          const bf16_t* __restrict__ Wt,  // [11][64 d][64 c] bf16 (transposed Wmix)
          const bf16_t* __restrict__ Wr,  // [11][12][64] bf16
          float* __restrict__ agg0, float* __restrict__ agg1, float* __restrict__ agg2,
          float* __restrict__ out0, float* __restrict__ out1, float* __restrict__ out2,
          float* __restrict__ enorm, int N, int E) {
  __shared__ WaveTile tiles[WAVES];
  const int lane = threadIdx.x & 31;
  const int wave = threadIdx.x >> 5;
  WaveTile& T = tiles[wave];
  const int el = lane & 15, half = lane >> 4;
  const int cb0 = half * 16, cb1 = 32 + half * 16;
  const int nTiles = (E + 15) >> 4;

  for (int tile = blockIdx.x * WAVES + wave; tile < nTiles; tile += gridDim.x * WAVES) {
    const int E0 = tile << 4;
    if (lane < 16) {
      int ge = min(E0 + lane, E - 1);
      T.uvec[lane][0] = rhat[ge * 3 + 0];
      T.uvec[lane][1] = rhat[ge * 3 + 1];
      T.uvec[lane][2] = rhat[ge * 3 + 2];
      T.eidx[lane] = idx_i[ge];
    }
    for (int it = lane; it < 16 * 12; it += 32) {
      int e = it / 12, b = it % 12;
      T.srbf[e][b] = rbf[(size_t)min(E0 + e, E - 1) * 12 + b];
    }
    __builtin_amdgcn_wave_barrier();
    { // prefetch next tile's rank-2 edge data into caches
      int nt = tile + gridDim.x * WAVES;
      if (nt < nTiles) __builtin_prefetch(edge2 + (size_t)(nt << 4) * 576 + lane * 144, 0, 1);
    }
    // ---- stage z-arrays (bf16, channel-contiguous) ----
#pragma unroll 1
    for (int it = 0; it < 32; ++it) {
      int idx = lane + (it << 5);
      int e = idx >> 6, c = idx & 63;
      int ge = min(E0 + e, E - 1);
      int j = min(max(idx_j[ge], 0), N - 1);
      float u0 = T.uvec[e][0], u1 = T.uvec[e][1], u2 = T.uvec[e][2];
      float b0 = node0[(size_t)j * 64 + c] + edge0[(size_t)ge * 64 + c];
      float b1[3], b2[9];
#pragma unroll
      for (int a = 0; a < 3; ++a)
        b1[a] = node1[((size_t)j * 64 + c) * 3 + a] + edge1[((size_t)ge * 64 + c) * 3 + a];
#pragma unroll
      for (int p = 0; p < 9; ++p)
        b2[p] = node2[((size_t)j * 64 + c) * 9 + p] + edge2[((size_t)ge * 64 + c) * 9 + p];
      float v2[3];
#pragma unroll
      for (int a = 0; a < 3; ++a) v2[a] = b2[a * 3 + 0] * u0 + b2[a * 3 + 1] * u1 + b2[a * 3 + 2] * u2;
      float s1 = b1[0] * u0 + b1[1] * u1 + b1[2] * u2;
      float s2 = v2[0] * u0 + v2[1] * u1 + v2[2] * u2;
      T.sb0[e][c] = (bf16_t)b0;
#pragma unroll
      for (int a = 0; a < 3; ++a) T.sb1[e][a][c] = (bf16_t)b1[a];
#pragma unroll
      for (int p = 0; p < 9; ++p) T.sb2[e][p][c] = (bf16_t)b2[p];
#pragma unroll
      for (int a = 0; a < 3; ++a) T.sv2[e][a][c] = (bf16_t)v2[a];
      T.ss1[e][c] = (bf16_t)s1;
      T.ss2[e][c] = (bf16_t)s2;
    }
    __builtin_amdgcn_wave_barrier();

    const float uu0 = T.uvec[el][0], uu1 = T.uvec[el][1], uu2 = T.uvec[el][2];
    const int gi = min(max(T.eidx[el], 0), N - 1);
    const bool valid = (E0 + el) < E;
    const size_t geL = (size_t)(E0 + el);
    const bf16_t* fb0 = T.fbuf[0][el];
    const bf16_t* fb1 = T.fbuf[1][el];
    const bf16_t* fb2 = T.fbuf[2][el];
    const bf16_t* fb3 = T.fbuf[3][el];

    // =============== o = 0 : combos k0..k2 ===============
#pragma unroll 1
    for (int g = 0; g < 3; ++g) {
      int k = g;
#pragma unroll 1
      for (int it = lane; it < 1024; it += 32) {
        int e = it >> 6, c = it & 63;
        float a0 = 0.0f;
#pragma unroll
        for (int b = 0; b < 12; ++b) a0 += T.srbf[e][b] * (float)Wr[(size_t)(k * 12 + b) * 64 + c];
        T.fbuf[g][e][c] = (bf16_t)a0;
      }
    }
    __builtin_amdgcn_wave_barrier();
    {
      v8f acc[4]; zero4(acc);
      gemmAcc(acc, Wt + 0 * 4096, mulFrag(T.sb0[el] + cb0, fb0 + cb0, 1.0f),
                                  mulFrag(T.sb0[el] + cb1, fb0 + cb1, 1.0f), lane);
      gemmAcc(acc, Wt + 1 * 4096, mulFrag(T.ss1[el] + cb0, fb1 + cb0, 1.0f),
                                  mulFrag(T.ss1[el] + cb1, fb1 + cb1, 1.0f), lane);
      gemmAcc(acc, Wt + 2 * 4096, mulFrag(T.ss2[el] + cb0, fb2 + cb0, 1.0f),
                                  mulFrag(T.ss2[el] + cb1, fb2 + cb1, 1.0f), lane);
#pragma unroll
      for (int t = 0; t < 4; ++t) {
#pragma unroll
        for (int r = 0; r < 8; ++r) {
          int d = t * 16 + r + 8 * half;
          float v = acc[t][r];
          if (MODE == 0) {
            if (valid) atomicAdd(agg0 + (size_t)gi * 64 + d, v * 0.0625f);
          } else {
            if (valid) out0[geL * 64 + d] = edge0[geL * 64 + d] + siluf(v);
          }
        }
      }
    }

    // =============== o = 1 : combos k3..k6 ===============
#pragma unroll 1
    for (int g = 0; g < 4; ++g) {
      int k = 3 + g;
#pragma unroll 1
      for (int it = lane; it < 1024; it += 32) {
        int e = it >> 6, c = it & 63;
        float a0 = 0.0f;
#pragma unroll
        for (int b = 0; b < 12; ++b) a0 += T.srbf[e][b] * (float)Wr[(size_t)(k * 12 + b) * 64 + c];
        T.fbuf[g][e][c] = (bf16_t)a0;
      }
    }
    __builtin_amdgcn_wave_barrier();
    {
      v8f nacc[4]; zero4(nacc);
#pragma unroll 1
      for (int a = 0; a < 3; ++a) {
        float ua = sel3(a, uu0, uu1, uu2);
        v8f acc[4]; zero4(acc);
        gemmAcc(acc, Wt + 3 * 4096, mulFrag(T.sb0[el] + cb0, fb0 + cb0, ua),
                                    mulFrag(T.sb0[el] + cb1, fb0 + cb1, ua), lane);
        gemmAcc(acc, Wt + 4 * 4096, mulFrag(T.sb1[el][a] + cb0, fb1 + cb0, 1.0f),
                                    mulFrag(T.sb1[el][a] + cb1, fb1 + cb1, 1.0f), lane);
        gemmAcc(acc, Wt + 5 * 4096, mulFrag(T.ss1[el] + cb0, fb2 + cb0, ua),
                                    mulFrag(T.ss1[el] + cb1, fb2 + cb1, ua), lane);
        gemmAcc(acc, Wt + 6 * 4096, mulFrag(T.sv2[el][a] + cb0, fb3 + cb0, 1.0f),
                                    mulFrag(T.sv2[el][a] + cb1, fb3 + cb1, 1.0f), lane);
#pragma unroll
        for (int t = 0; t < 4; ++t) {
#pragma unroll
          for (int r = 0; r < 8; ++r) {
            int d = t * 16 + r + 8 * half;
            float v = acc[t][r];
            if (MODE == 0) {
              if (valid) atomicAdd(agg1 + ((size_t)gi * 64 + d) * 3 + a, v * 0.0625f);
            } else {
              if (valid) out1[(geL * 64 + d) * 3 + a] = v;
              nacc[t][r] += v * v;
            }
          }
        }
      }
      if (MODE == 1 && valid) {
#pragma unroll
        for (int t = 0; t < 4; ++t) {
#pragma unroll
          for (int r = 0; r < 8; ++r) {
            int d = t * 16 + r + 8 * half;
            enorm[geL * 128 + d] = nacc[t][r];
          }
        }
      }
    }

    // =============== o = 2 : combos k7..k10 ===============
#pragma unroll 1
    for (int g = 0; g < 4; ++g) {
      int k = 7 + g;
#pragma unroll 1
      for (int it = lane; it < 1024; it += 32) {
        int e = it >> 6, c = it & 63;
        float a0 = 0.0f;
#pragma unroll
        for (int b = 0; b < 12; ++b) a0 += T.srbf[e][b] * (float)Wr[(size_t)(k * 12 + b) * 64 + c];
        T.fbuf[g][e][c] = (bf16_t)a0;
      }
    }
    __builtin_amdgcn_wave_barrier();
    {
      v8f nacc[4]; zero4(nacc);
#pragma unroll 1
      for (int s = 0; s < 9; ++s) {
        int a = s / 3;
        int b = s - a * 3;
        float ua = sel3(a, uu0, uu1, uu2);
        float ub = sel3(b, uu0, uu1, uu2);
        float sab = ua * ub;
        v8f acc[4]; zero4(acc);
        gemmAcc(acc, Wt + 7 * 4096,  mulFrag(T.sb0[el] + cb0, fb0 + cb0, sab),
                                     mulFrag(T.sb0[el] + cb1, fb0 + cb1, sab), lane);
        gemmAcc(acc, Wt + 8 * 4096,  mulFrag(T.sb1[el][a] + cb0, fb1 + cb0, ub),
                                     mulFrag(T.sb1[el][a] + cb1, fb1 + cb1, ub), lane);
        gemmAcc(acc, Wt + 9 * 4096,  mulFrag(T.sb2[el][s] + cb0, fb2 + cb0, 1.0f),
                                     mulFrag(T.sb2[el][s] + cb1, fb2 + cb1, 1.0f), lane);
        gemmAcc(acc, Wt + 10 * 4096, mulFrag(T.sv2[el][a] + cb0, fb3 + cb0, ub),
                                     mulFrag(T.sv2[el][a] + cb1, fb3 + cb1, ub), lane);
#pragma unroll
        for (int t = 0; t < 4; ++t) {
#pragma unroll
          for (int r = 0; r < 8; ++r) {
            int d = t * 16 + r + 8 * half;
            float v = acc[t][r];
            if (MODE == 0) {
              if (valid) atomicAdd(agg2 + ((size_t)gi * 64 + d) * 9 + s, v * 0.0625f);
            } else {
              if (valid) out2[(geL * 64 + d) * 9 + s] = v;
              nacc[t][r] += v * v;
            }
          }
        }
      }
      if (MODE == 1 && valid) {
#pragma unroll
        for (int t = 0; t < 4; ++t) {
#pragma unroll
          for (int r = 0; r < 8; ++r) {
            int d = t * 16 + r + 8 * half;
            enorm[geL * 128 + 64 + d] = nacc[t][r];
          }
        }
      }
    }
  }
}

// ---------------------------------------------------------------------------
// gate B-fragment from 16-contiguous f32 norm runs
// ---------------------------------------------------------------------------
__device__ __forceinline__ void gateFrag(v16bf* B, const float* __restrict__ norm,
                                         size_t rowBase, int wOff, int half) {
#pragma unroll
  for (int kc = 0; kc < 2; ++kc) {
    v16bf bfr;
#pragma unroll
    for (int q = 0; q < 16; ++q) {
      int c = kc * 32 + half * 16 + q;
      bfr[q] = (bf16_t)norm[rowBase + wOff + c];
    }
    B[kc] = bfr;
  }
}

// node update: node' = node + nonlinear(agg) (norms precomputed in nnorm)
__global__ VGPR_CAP __launch_bounds__(128)
void kNodeUpdate(const float* __restrict__ n0, const float* __restrict__ n1,
                 const float* __restrict__ n2, const float* __restrict__ agg0,
                 const float* __restrict__ agg1, const float* __restrict__ agg2,
                 const float* __restrict__ nnorm, const bf16_t* __restrict__ WgT,
                 const float* __restrict__ bg, float* __restrict__ o0,
                 float* __restrict__ o1, float* __restrict__ o2, int N) {
  int lane = threadIdx.x & 31, wave = threadIdx.x >> 5;
  int at = lane & 15, half = lane >> 4;
  int tiles = (N + 15) >> 4;
  for (int tile = blockIdx.x * 4 + wave; tile < tiles; tile += gridDim.x * 4) {
    int A0 = tile << 4;
    int ga = min(A0 + at, N - 1);
    bool valid = (A0 + at) < N;
    size_t rowBase = (size_t)ga * 128;
    // o = 0 path (elementwise silu)
    if (valid) {
#pragma unroll
      for (int t = 0; t < 4; ++t) {
#pragma unroll
        for (int r = 0; r < 8; ++r) {
          int d = t * 16 + r + 8 * half;
          size_t i0 = (size_t)ga * 64 + d;
          o0[i0] = n0[i0] + siluf(agg0[i0]);
        }
      }
    }
    // gated paths, w sequential to keep gate liveness short
#pragma unroll 1
    for (int w = 0; w < 2; ++w) {
      v16bf B[2];
      gateFrag(B, nnorm, rowBase, w * 64, half);
      v8f acc[4]; zero4(acc);
      gemmAcc(acc, WgT + w * 4096, B[0], B[1], lane);
      if (valid) {
#pragma unroll
        for (int t = 0; t < 4; ++t) {
#pragma unroll
          for (int r = 0; r < 8; ++r) {
            int d = t * 16 + r + 8 * half;
            float gate = siluf(acc[t][r] + bg[w * 64 + d]);
            size_t i0 = (size_t)ga * 64 + d;
            if (w == 0) {
#pragma unroll
              for (int s = 0; s < 3; ++s) { size_t i = i0 * 3 + s; o1[i] = n1[i] + agg1[i] * gate; }
            } else {
#pragma unroll
              for (int s = 0; s < 9; ++s) { size_t i = i0 * 9 + s; o2[i] = n2[i] + agg2[i] * gate; }
            }
          }
        }
      }
    }
  }
}

// edge finalize: edge' = edge + msg_raw * silu(norm@Wg + bg)
__global__ VGPR_CAP __launch_bounds__(128)
void kEdgeFinal(const float* __restrict__ e1, const float* __restrict__ e2,
                const float* __restrict__ enorm, const bf16_t* __restrict__ WgT,
                const float* __restrict__ bg, float* __restrict__ o1,
                float* __restrict__ o2, int E) {
  int lane = threadIdx.x & 31, wave = threadIdx.x >> 5;
  int el = lane & 15, half = lane >> 4;
  int tiles = (E + 15) >> 4;
  for (int tile = blockIdx.x * 4 + wave; tile < tiles; tile += gridDim.x * 4) {
    int E0 = tile << 4;
    int ge = min(E0 + el, E - 1);
    bool valid = (E0 + el) < E;
    size_t rowBase = (size_t)ge * 128;
#pragma unroll 1
    for (int w = 0; w < 2; ++w) {
      v16bf B[2];
      gateFrag(B, enorm, rowBase, w * 64, half);
      v8f acc[4]; zero4(acc);
      gemmAcc(acc, WgT + w * 4096, B[0], B[1], lane);
      if (valid) {
#pragma unroll
        for (int t = 0; t < 4; ++t) {
#pragma unroll
          for (int r = 0; r < 8; ++r) {
            int d = t * 16 + r + 8 * half;
            float gate = siluf(acc[t][r] + bg[w * 64 + d]);
            size_t i0 = (size_t)ge * 64 + d;
            if (w == 0) {
#pragma unroll
              for (int s = 0; s < 3; ++s) { size_t i = i0 * 3 + s; float raw = o1[i]; o1[i] = e1[i] + raw * gate; }
            } else {
#pragma unroll
              for (int s = 0; s < 9; ++s) { size_t i = i0 * 9 + s; float raw = o2[i]; o2[i] = e2[i] + raw * gate; }
            }
          }
        }
      }
    }
  }
}

// ---------------------------------------------------------------------------
extern "C" void kernel_launch(void* const* d_in, const int* in_sizes, int n_in,
                              void* d_out, int out_size, void* d_ws, size_t ws_size,
                              hipStream_t stream) {
  const float* node0 = (const float*)d_in[0];
  const float* node1 = (const float*)d_in[1];
  const float* node2 = (const float*)d_in[2];
  const float* edge0 = (const float*)d_in[3];
  const float* edge1 = (const float*)d_in[4];
  const float* edge2 = (const float*)d_in[5];
  const int* idx_i   = (const int*)d_in[6];
  const int* idx_j   = (const int*)d_in[7];
  const float* rij   = (const float*)d_in[8];
  const float* Wrbf_n = (const float*)d_in[9];
  const float* Wmix_n = (const float*)d_in[10];
  const float* Wg_n   = (const float*)d_in[11];
  const float* bg_n   = (const float*)d_in[12];
  const float* Wrbf_e = (const float*)d_in[13];
  const float* Wmix_e = (const float*)d_in[14];
  const float* Wg_e   = (const float*)d_in[15];
  const float* bg_e   = (const float*)d_in[16];
  const int N = in_sizes[0] / 64;
  const int E = in_sizes[6];

  char* wptr = (char*)d_ws;
  auto take = [&](size_t bytes) -> char* {
    char* p = wptr;
    wptr += (bytes + 255) & ~(size_t)255;
    return p;
  };
  float* rbf   = (float*)take((size_t)E * 12 * 4);
  float* rhat  = (float*)take((size_t)E * 3 * 4);
  float* agg0  = (float*)take((size_t)N * 64 * 4);
  float* agg1  = (float*)take((size_t)N * 192 * 4);
  float* agg2  = (float*)take((size_t)N * 576 * 4);
  float* nnorm = (float*)take((size_t)N * 128 * 4);
  float* enorm = (float*)take((size_t)E * 128 * 4);
  bf16_t* WtN  = (bf16_t*)take((size_t)11 * 4096 * 2);
  bf16_t* WtE  = (bf16_t*)take((size_t)11 * 4096 * 2);
  bf16_t* WrN  = (bf16_t*)take((size_t)11 * 768 * 2);
  bf16_t* WrE  = (bf16_t*)take((size_t)11 * 768 * 2);
  bf16_t* WgNT = (bf16_t*)take((size_t)2 * 4096 * 2);
  bf16_t* WgET = (bf16_t*)take((size_t)2 * 4096 * 2);

  float* o = (float*)d_out;
  float* on0 = o;
  float* on1 = on0 + (size_t)N * 64;
  float* on2 = on1 + (size_t)N * 192;
  float* oe0 = on2 + (size_t)N * 576;
  float* oe1 = oe0 + (size_t)E * 64;
  float* oe2 = oe1 + (size_t)E * 192;

  kZero<<<(int)(((size_t)N * 64 + 255) / 256), 256, 0, stream>>>(agg0, (long)N * 64);
  kZero<<<(int)(((size_t)N * 192 + 255) / 256), 256, 0, stream>>>(agg1, (long)N * 192);
  kZero<<<(int)(((size_t)N * 576 + 255) / 256), 256, 0, stream>>>(agg2, (long)N * 576);
  kPrepEdge<<<(E + 255) / 256, 256, 0, stream>>>(rij, rbf, rhat, E);
  kCvtT<<<(11 * 4096 + 255) / 256, 256, 0, stream>>>(Wmix_n, WtN, 11);
  kCvtT<<<(11 * 4096 + 255) / 256, 256, 0, stream>>>(Wmix_e, WtE, 11);
  kCvt<<<(11 * 768 + 255) / 256, 256, 0, stream>>>(Wrbf_n, WrN, 11 * 768);
  kCvt<<<(11 * 768 + 255) / 256, 256, 0, stream>>>(Wrbf_e, WrE, 11 * 768);
  kCvtT<<<(2 * 4096 + 255) / 256, 256, 0, stream>>>(Wg_n, WgNT, 2);
  kCvtT<<<(2 * 4096 + 255) / 256, 256, 0, stream>>>(Wg_e, WgET, 2);

  // conv 1: node messages -> atomic segment sum into agg (scaled 1/16)
  kMsg<0><<<512, BLK, 0, stream>>>(node0, node1, node2, edge0, edge1, edge2, idx_i, idx_j,
                                   rbf, rhat, WtN, WrN, agg0, agg1, agg2,
                                   nullptr, nullptr, nullptr, nullptr, N, E);
  // node gate-input norms (streaming, coalesced)
  kNodeNorm<<<(N * 64 + 255) / 256, 256, 0, stream>>>(agg1, agg2, nnorm, N);
  // node' = node + nonlinear(agg)
  kNodeUpdate<<<160, 128, 0, stream>>>(node0, node1, node2, agg0, agg1, agg2, nnorm,
                                       WgNT, bg_n, on0, on1, on2, N);
  // conv 2: edge messages using updated nodes; raw msgs + norms (o=0 final)
  kMsg<1><<<512, BLK, 0, stream>>>(on0, on1, on2, edge0, edge1, edge2, idx_i, idx_j,
                                   rbf, rhat, WtE, WrE, nullptr, nullptr, nullptr,
                                   oe0, oe1, oe2, enorm, N, E);
  // edge' = edge + msg * silu(norm@Wg + bg)
  kEdgeFinal<<<512, 128, 0, stream>>>(edge1, edge2, enorm, WgET, bg_e, oe1, oe2, E);
}
